// DeformConv3dPack_74483322847780
// MI455X (gfx1250) — matrix operand order repaired and checked by reading, not compiled, not run on hardware
//
#include <hip/hip_runtime.h>

// ---------------------------------------------------------------------------
// Deformable Conv3d (B=1, C=64, D=8, H=64, W=64, O=64, 3x3x3, s=1, p=1, d=1)
// MI455X / gfx1250: both convs as implicit GEMM on v_wmma_f32_16x16x32_f16.
// Rev 2: double-buffered (software-pipelined) K-loop with #pragma unroll 2 to
// stop the compiler from caching the whole B panel in 432 VGPRs; balanced
// 6-wave workgroup for the 96-row offset conv.
// ---------------------------------------------------------------------------

typedef __attribute__((ext_vector_type(16))) _Float16 v16h;
typedef __attribute__((ext_vector_type(8)))  _Float16 v8h;
typedef __attribute__((ext_vector_type(8)))  float    v8f;

#define C_IN      64
#define D_IN      8
#define H_IN      64
#define W_IN      64
#define P_TOT     32768          // D*H*W output points (stride1/pad1)
#define K_TAPS    27
#define KG        1728           // GEMM K = C_IN * K_TAPS
#define O_MAIN    64
#define O_OFF     81             // 3*27 offset channels
#define O_OFF_PAD 96             // padded to a multiple of 16 for WMMA tiles
#define TILE_N    16             // output positions per workgroup

// d_ws byte offsets (all 256B aligned), ~14.7 MB total
#define WS_OFF    0ull                          // f32 [81][32768]  = 10,616,832 B
#define WS_WF16   10616832ull                   // f16 [64][1728]   =    221,184 B
#define WS_WOFF16 10838016ull                   // f16 [96][1728]   =    331,776 B
#define WS_XF16   11169792ull                   // f16 [64][32768]  =  4,194,304 B

// ------------------------------ prep kernels -------------------------------

__global__ void k_cvt_f16(const float* __restrict__ src,
                          _Float16* __restrict__ dst, int n) {
  int i = blockIdx.x * blockDim.x + threadIdx.x;
  if (i < n) dst[i] = (_Float16)src[i];
}

__global__ void k_pad_woff(const float* __restrict__ src,
                           _Float16* __restrict__ dst) {
  int i = blockIdx.x * blockDim.x + threadIdx.x;   // over 96*1728
  if (i >= O_OFF_PAD * KG) return;
  dst[i] = (i < O_OFF * KG) ? (_Float16)src[i] : (_Float16)0.f;
}

// -------------------------- WMMA fragment helpers --------------------------
// A: 16x32 f16.  lane L holds row M = L&15; K = kk + (L>>4)*16 .. +15,
//    consecutive K packed in VGPRs -> two 16B row loads.
// B: 32x16 f16.  lane L holds col N = L&15; rows K = kk + (L>>4)*16 .. +15,
//    consecutive K packed in VGPRs -> two ds_load_b128 from the swizzled LDS
//    panel stored as [K/16 block][n][16 consecutive K halfwords].

__device__ __forceinline__ v16h make_v16h(v8h lo, v8h hi) {
  v16h r;
#pragma unroll
  for (int e = 0; e < 8; ++e) { r[e] = lo[e]; r[8 + e] = hi[e]; }
  return r;
}

__device__ __forceinline__ v16h load_a_frag(const _Float16* __restrict__ Arow,
                                            int kk, int khalf) {
  v8h lo = *(const v8h*)(Arow + kk + khalf * 16);
  v8h hi = *(const v8h*)(Arow + kk + khalf * 16 + 8);
  return make_v16h(lo, hi);
}

__device__ __forceinline__ v16h load_b_frag(const _Float16* __restrict__ Bt,
                                            int kk, int khalf, int n) {
  const _Float16* src = Bt + ((((kk >> 4) + khalf) * TILE_N + n) << 4);
  v8h lo = *(const v8h*)(src);
  v8h hi = *(const v8h*)(src + 8);
  return make_v16h(lo, hi);
}

// Software-pipelined 16x16 GEMM tile over K=1728: next-iteration A (global,
// L2-hot) and B (LDS) fragments are issued before the current WMMA, with a
// branchless wrap so loads are unconditional.  #pragma unroll 2 keeps live
// registers to {acc + 2 frags each} instead of the whole 432-VGPR B panel.
__device__ __forceinline__ v8f gemm_tile(const _Float16* __restrict__ Arow,
                                         const _Float16* __restrict__ Bt,
                                         int khalf, int n) {
  v8f acc = {};
  v16h a_cur = load_a_frag(Arow, 0, khalf);
  v16h b_cur = load_b_frag(Bt, 0, khalf, n);
#pragma unroll 2
  for (int kk = 0; kk < KG; kk += 32) {
    const int kn = (kk + 32 == KG) ? 0 : kk + 32;   // harmless wrap prefetch
    v16h a_nxt = load_a_frag(Arow, kn, khalf);
    v16h b_nxt = load_b_frag(Bt, kn, khalf, n);
    acc = __builtin_amdgcn_wmma_f32_16x16x32_f16(
        false, a_cur, false, b_cur, (short)0, acc, false, false);
    a_cur = a_nxt;
    b_cur = b_nxt;
  }
  return acc;
}

// ------------------- kernel A: offset conv (implicit GEMM) -----------------
// One workgroup (6 waves, 192 threads) per 16 output positions; LDS holds the
// 1728x16 zero-padded im2col panel; each wave owns one of the 6 M-tiles of
// the padded 96 offset channels.

__global__ __launch_bounds__(192) void k_offset_conv(
    const _Float16* __restrict__ xf16,
    const _Float16* __restrict__ woff16,
    const float* __restrict__ boff,
    float* __restrict__ off) {
  __shared__ _Float16 Bt[KG * TILE_N];          // 55,296 B
  const int tile = blockIdx.x;
  const int tid  = threadIdx.x;
  const int p0   = tile * TILE_N;

  // phase 1: build swizzled im2col panel
  for (int idx = tid; idx < KG * TILE_N; idx += 192) {
    int k = idx >> 4, n = idx & 15;
    int c = k / K_TAPS, tap = k - c * K_TAPS;
    int kd = tap / 9, r = tap - kd * 9, kh = r / 3, kw = r - kh * 3;
    int p  = p0 + n;
    int pz = p >> 12, py = (p >> 6) & 63, px = p & 63;
    int z = pz + kd - 1, y = py + kh - 1, xx = px + kw - 1;
    _Float16 v = (_Float16)0.f;
    if ((unsigned)z < D_IN && (unsigned)y < H_IN && (unsigned)xx < W_IN)
      v = xf16[c * P_TOT + (z * H_IN + y) * W_IN + xx];
    Bt[(((k >> 4) * TILE_N + n) << 4) + (k & 15)] = v;
  }
  __syncthreads();

  const int lane = tid & 31, mt = tid >> 5;     // wave == M-tile, 0..5
  const int n = lane & 15, khalf = lane >> 4;
  const int p = p0 + n;

  const _Float16* Arow = woff16 + (mt * 16 + n) * KG;  // row M = lane&15
  v8f acc = gemm_tile(Arow, Bt, khalf, n);

#pragma unroll
  for (int rr = 0; rr < 8; ++rr) {
    int oc = mt * 16 + khalf * 8 + rr;   // C layout: M = r + 8*(lane>>4)
    if (oc < O_OFF) off[oc * P_TOT + p] = acc[rr] + boff[oc];
  }
}

// --------------- kernel B: trilinear gather + main conv GEMM ---------------

__global__ __launch_bounds__(128) void k_deform_gemm(
    const float* __restrict__ x,
    const float* __restrict__ off,
    const _Float16* __restrict__ w16,
    const float* __restrict__ bias,
    float* __restrict__ out) {
  __shared__ _Float16 Bt[KG * TILE_N];
  const int tile = blockIdx.x;
  const int tid  = threadIdx.x;
  const int p0   = tile * TILE_N;

  // phase 1: 27 taps x 16 positions; corner weights shared across channels
  for (int pr = tid; pr < K_TAPS * TILE_N; pr += 128) {
    int tap = pr >> 4, n = pr & 15;
    int kd = tap / 9, r2 = tap - kd * 9, kh = r2 / 3, kw = r2 - kh * 3;
    int p  = p0 + n;
    int pz = p >> 12, py = (p >> 6) & 63, px = p & 63;
    float z  = (float)(pz + kd - 1) + off[(tap * 3 + 0) * P_TOT + p];
    float y  = (float)(py + kh - 1) + off[(tap * 3 + 1) * P_TOT + p];
    float w_ = (float)(px + kw - 1) + off[(tap * 3 + 2) * P_TOT + p];
    float z0f = floorf(z), y0f = floorf(y), x0f = floorf(w_);
    int z0 = (int)z0f, y0 = (int)y0f, x0 = (int)x0f;
    float fz = z - z0f, fy = y - y0f, fx = w_ - x0f;
    float wz[2] = {1.f - fz, fz}, wy[2] = {1.f - fy, fy}, wx[2] = {1.f - fx, fx};
    int cidx[8]; float cw[8];
#pragma unroll
    for (int q = 0; q < 8; ++q) {
      int dz = q >> 2, dy = (q >> 1) & 1, dx = q & 1;
      int zi = z0 + dz, yi = y0 + dy, xi = x0 + dx;
      bool ok = ((unsigned)zi < D_IN) && ((unsigned)yi < H_IN) &&
                ((unsigned)xi < W_IN);
      int zc = zi < 0 ? 0 : (zi > D_IN - 1 ? D_IN - 1 : zi);
      int yc = yi < 0 ? 0 : (yi > H_IN - 1 ? H_IN - 1 : yi);
      int xc = xi < 0 ? 0 : (xi > W_IN - 1 ? W_IN - 1 : xi);
      cidx[q] = (zc * H_IN + yc) * W_IN + xc;
      cw[q]   = ok ? wz[dz] * wy[dy] * wx[dx] : 0.f;
    }
    for (int c = 0; c < C_IN; ++c) {
      const float* xc = x + c * P_TOT;
      float v = 0.f;
#pragma unroll
      for (int q = 0; q < 8; ++q) v += cw[q] * xc[cidx[q]];
      int k = c * K_TAPS + tap;
      Bt[(((k >> 4) * TILE_N + n) << 4) + (k & 15)] = (_Float16)v;
    }
  }
  __syncthreads();

  const int lane = tid & 31, mt = tid >> 5;    // M=64 -> 4 tiles, 1 per wave
  const int n = lane & 15, khalf = lane >> 4;
  const int p = p0 + n;

  const _Float16* Arow = w16 + (mt * 16 + n) * KG;
  v8f acc = gemm_tile(Arow, Bt, khalf, n);

#pragma unroll
  for (int rr = 0; rr < 8; ++rr) {
    int oc = mt * 16 + khalf * 8 + rr;
    out[oc * P_TOT + p] = acc[rr] + bias[oc];
  }
}

// ------------------------------- launcher ----------------------------------

extern "C" void kernel_launch(void* const* d_in, const int* in_sizes, int n_in,
                              void* d_out, int out_size, void* d_ws,
                              size_t ws_size, hipStream_t stream) {
  (void)in_sizes; (void)n_in; (void)out_size; (void)ws_size;
  const float* x      = (const float*)d_in[0];
  const float* w_off  = (const float*)d_in[1];
  const float* b_off  = (const float*)d_in[2];
  const float* weight = (const float*)d_in[3];
  const float* bias   = (const float*)d_in[4];
  float* out = (float*)d_out;

  char* ws = (char*)d_ws;
  float*    offbuf = (float*)(ws + WS_OFF);
  _Float16* w16    = (_Float16*)(ws + WS_WF16);
  _Float16* woff16 = (_Float16*)(ws + WS_WOFF16);
  _Float16* xf16   = (_Float16*)(ws + WS_XF16);

  k_cvt_f16<<<(C_IN * P_TOT + 255) / 256, 256, 0, stream>>>(x, xf16,
                                                            C_IN * P_TOT);
  k_cvt_f16<<<(O_MAIN * KG + 255) / 256, 256, 0, stream>>>(weight, w16,
                                                           O_MAIN * KG);
  k_pad_woff<<<(O_OFF_PAD * KG + 255) / 256, 256, 0, stream>>>(w_off, woff16);

  k_offset_conv<<<P_TOT / TILE_N, 192, 0, stream>>>(xf16, woff16, b_off,
                                                    offbuf);
  k_deform_gemm<<<P_TOT / TILE_N, 128, 0, stream>>>(x, offbuf, w16, bias, out);
}
